// model_52089363366199
// MI455X (gfx1250) — compile-verified, matrix operand order checked by reading, not run on hardware
//
#include <hip/hip_runtime.h>

typedef __attribute__((ext_vector_type(16))) __bf16 v16bf;
typedef __attribute__((ext_vector_type(8)))  float  v8f;

__device__ __forceinline__ void atomic_add_f32(float* p, float v) {
  (void)__hip_atomic_fetch_add(p, v, __ATOMIC_RELAXED, __HIP_MEMORY_SCOPE_AGENT);
}

// ---------------- degree / invdeg ----------------
__global__ void deg_kernel(const int* __restrict__ dst, float* __restrict__ deg, int E) {
  int i = blockIdx.x * blockDim.x + threadIdx.x;
  if (i < E) atomic_add_f32(&deg[dst[i]], 1.0f);
}

__global__ void invdeg_kernel(const float* __restrict__ deg, float* __restrict__ invdeg, int N) {
  int i = blockIdx.x * blockDim.x + threadIdx.x;
  if (i < N) invdeg[i] = 1.0f / fmaxf(deg[i], 1.0f);
}

// ---------------- scatter-add: one wave per edge, float4 per lane ----------------
__global__ void scatter_kernel(const float* __restrict__ feat, const int* __restrict__ src,
                               const int* __restrict__ dst, float* __restrict__ agg, int E) {
  int wid  = (blockIdx.x * blockDim.x + threadIdx.x) >> 5;
  int lane = threadIdx.x & 31;
  if (wid >= E) return;
  int s = src[wid];
  int d = dst[wid];
  const float4 v = ((const float4*)(feat + (size_t)s * 128))[lane];
  float* a = agg + (size_t)d * 128 + lane * 4;
  atomic_add_f32(a + 0, v.x);
  atomic_add_f32(a + 1, v.y);
  atomic_add_f32(a + 2, v.z);
  atomic_add_f32(a + 3, v.w);
}

// ---------------- weight packing into WMMA B-fragment layout (bf16 hi/lo) ------------
// Fragment index: [ncol-tile nt (8)][k-step kt (8 over concat K=256)][lane (32)][16 bf16]
// Lane l holds column n = nt*16 + (l&15); half = l>>4 selects K groups
// {0..7,16..23} (half=0) / {8..15,24..31} (half=1) within the 32-wide k-step.
__global__ void pack_w_kernel(const float* __restrict__ Wl, const float* __restrict__ Wr,
                              __bf16* __restrict__ outHi, __bf16* __restrict__ outLo) {
  int t    = blockIdx.x * blockDim.x + threadIdx.x;   // 0..2047
  int lane = t & 31;
  int kt   = (t >> 5) & 7;
  int nt   = t >> 8;
  int half = lane >> 4;
  int n    = nt * 16 + (lane & 15);
  int kk   = (kt & 3) * 32;
  const float* Wrow = ((kt < 4) ? Wl : Wr) + (size_t)n * 128;
  __bf16* hi = outHi + ((size_t)t) * 16;
  __bf16* lo = outLo + ((size_t)t) * 16;
  #pragma unroll
  for (int j = 0; j < 8; ++j) {
    float f0 = Wrow[kk + half * 8 + j];
    float f1 = Wrow[kk + 16 + half * 8 + j];
    __bf16 h0 = (__bf16)f0;
    __bf16 h1 = (__bf16)f1;
    hi[j]     = h0;  lo[j]     = (__bf16)(f0 - (float)h0);
    hi[8 + j] = h1;  lo[8 + j] = (__bf16)(f1 - (float)h1);
  }
}

// ---------------- fused SAGE GEMM: out = relu([agg*invdeg | x] @ [Wl;Wr]^T + b) ----------
// Block = 256 threads = 8 waves. Wave w -> 16x16 tile at (blockIdx.x*16, w*16).
// Split-bf16 (hi/lo) emulation of fp32 matmul: 3 WMMAs per 32-wide k-step.
__global__ void __launch_bounds__(256)
sage_gemm_kernel(const float* __restrict__ agg, const float* __restrict__ invdeg,
                 const float* __restrict__ xin,
                 const __bf16* __restrict__ Bh, const __bf16* __restrict__ Bl,
                 const float* __restrict__ bias,
                 float* __restrict__ out, int do_relu) {
  const int wave = threadIdx.x >> 5;       // 0..7 -> column tile
  const int lane = threadIdx.x & 31;
  const int half = lane >> 4;
  const int m    = lane & 15;              // row within tile (A) / col within tile (B)
  const int row0 = blockIdx.x * 16;

  const float s = invdeg[row0 + m];
  const float* Arow0 = agg + (size_t)(row0 + m) * 128;   // scaled half of K
  const float* Arow1 = xin + (size_t)(row0 + m) * 128;   // self half of K

  const v16bf* BhP = (const v16bf*)Bh + (size_t)wave * 8 * 32 + lane;
  const v16bf* BlP = (const v16bf*)Bl + (size_t)wave * 8 * 32 + lane;

  v8f acc = {};
  #pragma unroll 1
  for (int kt = 0; kt < 8; ++kt) {
    const bool  first = (kt < 4);
    const int   kk    = (first ? kt : (kt - 4)) * 32;
    const float asc   = first ? s : 1.0f;
    const float* Arow = first ? Arow0 : Arow1;

    float4 a0 = *(const float4*)(Arow + kk + half * 8);
    float4 a1 = *(const float4*)(Arow + kk + half * 8 + 4);
    float4 a2 = *(const float4*)(Arow + kk + 16 + half * 8);
    float4 a3 = *(const float4*)(Arow + kk + 16 + half * 8 + 4);

    float a[16] = {a0.x, a0.y, a0.z, a0.w, a1.x, a1.y, a1.z, a1.w,
                   a2.x, a2.y, a2.z, a2.w, a3.x, a3.y, a3.z, a3.w};

    v16bf ah, al;
    #pragma unroll
    for (int j = 0; j < 16; ++j) {
      float f  = a[j] * asc;
      __bf16 h = (__bf16)f;
      ah[j] = h;
      al[j] = (__bf16)(f - (float)h);
    }

    v16bf bh = BhP[kt * 32];
    v16bf bl = BlP[kt * 32];

    acc = __builtin_amdgcn_wmma_f32_16x16x32_bf16(false, ah, false, bh, (short)0, acc, false, false);
    acc = __builtin_amdgcn_wmma_f32_16x16x32_bf16(false, ah, false, bl, (short)0, acc, false, false);
    acc = __builtin_amdgcn_wmma_f32_16x16x32_bf16(false, al, false, bh, (short)0, acc, false, false);
  }

  const int n    = wave * 16 + m;
  const float bv = bias[n];
  #pragma unroll
  for (int r = 0; r < 8; ++r) {
    int row = row0 + r + 8 * half;         // D layout: VGPR r -> M=r (lanes 0-15), M=r+8 (16-31)
    float v = acc[r] + bv;
    if (do_relu) v = fmaxf(v, 0.0f);
    out[(size_t)row * 128 + n] = v;
  }
}

// ---------------- score: one wave per node, dot(emb[i], Wfc) + bfc ----------------
__global__ void score_kernel(const float* __restrict__ emb, const float* __restrict__ Wfc,
                             const float* __restrict__ bfc, float* __restrict__ score, int N) {
  int wid  = (blockIdx.x * blockDim.x + threadIdx.x) >> 5;
  int lane = threadIdx.x & 31;
  if (wid >= N) return;
  const float4 v = ((const float4*)(emb + (size_t)wid * 128))[lane];
  const float4 w = ((const float4*)Wfc)[lane];
  float p = v.x * w.x + v.y * w.y + v.z * w.z + v.w * w.w;
  #pragma unroll
  for (int off = 16; off > 0; off >>= 1) p += __shfl_xor(p, off, 32);
  if (lane == 0) score[wid] = p + bfc[0];
}

extern "C" void kernel_launch(void* const* d_in, const int* in_sizes, int n_in,
                              void* d_out, int out_size, void* d_ws, size_t ws_size,
                              hipStream_t stream) {
  const float* x   = (const float*)d_in[0];
  const int*   ei  = (const int*)d_in[1];
  const float* W1l = (const float*)d_in[2];
  const float* b1  = (const float*)d_in[3];
  const float* W1r = (const float*)d_in[4];
  const float* W2l = (const float*)d_in[5];
  const float* b2  = (const float*)d_in[6];
  const float* W2r = (const float*)d_in[7];
  const float* Wfc = (const float*)d_in[8];
  const float* bfc = (const float*)d_in[9];

  const int N = in_sizes[0] / 128;   // 100000
  const int E = in_sizes[1] / 2;     // 1600000
  const int* src = ei;
  const int* dst = ei + E;

  float* score = (float*)d_out;            // [N]
  float* emb   = (float*)d_out + N;        // [N,128]

  const size_t fragElems = (size_t)8 * 8 * 32 * 16;   // 32768 bf16 per matrix pack

  char* w = (char*)d_ws;
  float*  deg    = (float*)w;   w += (size_t)N * 4;
  float*  invdeg = (float*)w;   w += (size_t)N * 4;
  float*  agg    = (float*)w;   w += (size_t)N * 128 * 4;
  float*  h1     = (float*)w;   w += (size_t)N * 128 * 4;
  __bf16* w1Hi   = (__bf16*)w;  w += fragElems * 2;
  __bf16* w1Lo   = (__bf16*)w;  w += fragElems * 2;
  __bf16* w2Hi   = (__bf16*)w;  w += fragElems * 2;
  __bf16* w2Lo   = (__bf16*)w;

  const size_t aggBytes = (size_t)N * 128 * 4;

  // ---- weight packing (tiny) + degree (shared by both layers) ----
  pack_w_kernel<<<8, 256, 0, stream>>>(W1l, W1r, w1Hi, w1Lo);
  pack_w_kernel<<<8, 256, 0, stream>>>(W2l, W2r, w2Hi, w2Lo);
  hipMemsetAsync(deg, 0, (size_t)N * 4, stream);
  deg_kernel<<<(E + 255) / 256, 256, 0, stream>>>(dst, deg, E);
  invdeg_kernel<<<(N + 255) / 256, 256, 0, stream>>>(deg, invdeg, N);

  // ---- layer 1 ----
  hipMemsetAsync(agg, 0, aggBytes, stream);
  scatter_kernel<<<(E + 7) / 8, 256, 0, stream>>>(x, src, dst, agg, E);
  sage_gemm_kernel<<<N / 16, 256, 0, stream>>>(agg, invdeg, x, w1Hi, w1Lo, b1, h1, 1);

  // ---- layer 2 (emb written straight into output) ----
  hipMemsetAsync(agg, 0, aggBytes, stream);
  scatter_kernel<<<(E + 7) / 8, 256, 0, stream>>>(h1, src, dst, agg, E);
  sage_gemm_kernel<<<N / 16, 256, 0, stream>>>(agg, invdeg, h1, w2Hi, w2Lo, b2, emb, 1);

  // ---- score head ----
  score_kernel<<<(N + 7) / 8, 256, 0, stream>>>(emb, Wfc, bfc, score, N);
}
